// PyramidalAttention_42631845380150
// MI455X (gfx1250) — matrix operand order repaired and checked by reading, not compiled, not run on hardware
//
#include <hip/hip_runtime.h>

#define BB 4
#define SS 1024
#define DD 512
#define HH 8
#define HDm 64

typedef __attribute__((ext_vector_type(16))) _Float16 v16h;
typedef __attribute__((ext_vector_type(8)))  _Float16 v8h;
typedef __attribute__((ext_vector_type(8)))  float    v8f;

union AV { v16h v; v8h h[2]; };

#define WMMA_F16(a, b, c) \
  __builtin_amdgcn_wmma_f32_16x16x32_f16(false, (a), false, (b), (short)0, (c), false, false)

// A-matrix 16x32 f16 fragment from row-major src (ld = row stride in elems).
// ISA layout: lanes 0-15 (M=lane): K 0-7 then 16-23; lanes 16-31 (M=lane-16): K 8-15 then 24-31.
static __device__ inline v16h load_A16x32(const _Float16* __restrict__ src, int ld, int lane) {
  const int g = lane >> 4, m = lane & 15;
  const _Float16* p = src + (size_t)m * ld + 8 * g;
  AV av;
  av.h[0] = *(const v8h*)(p);        // K = 8g .. 8g+7
  av.h[1] = *(const v8h*)(p + 16);   // K = 16+8g .. 16+8g+7
  return av.v;
}

// B-matrix 32x16 f16 fragment. srcT is B transposed ([N][K] row-major, ld = K stride).
// Layout: lane%16 = N column; lanes 0-15 hold K 0-7 then 16-23, lanes 16-31 hold K 8-15 then 24-31.
static __device__ inline v16h load_BT32x16(const _Float16* __restrict__ srcT, int ld, int lane) {
  const int g = lane >> 4, n = lane & 15;
  const _Float16* p = srcT + (size_t)n * ld + 8 * g;
  AV av;
  av.h[0] = *(const v8h*)(p);
  av.h[1] = *(const v8h*)(p + 16);
  return av.v;
}

// ---------------- conversion kernels ----------------
__global__ void cvt_x_kernel(const float* __restrict__ x, _Float16* __restrict__ xh, int n) {
  for (int i = blockIdx.x * blockDim.x + threadIdx.x; i < n; i += gridDim.x * blockDim.x)
    xh[i] = (_Float16)x[i];
}

// w [H][D][HD] f32 -> wt [H][HD][D] f16 (transpose inner dims)
__global__ void cvt_w_kernel(const float* __restrict__ w, _Float16* __restrict__ wt) {
  const int n = HH * DD * HDm;
  for (int i = blockIdx.x * blockDim.x + threadIdx.x; i < n; i += gridDim.x * blockDim.x) {
    int h = i / (DD * HDm);
    int rem = i % (DD * HDm);
    int d = rem / HDm;
    int e = rem % HDm;
    wt[((size_t)h * HDm + e) * DD + d] = (_Float16)w[i];
  }
}

// wo [D][D] f32 -> woh [D][D] f16 transposed (woh[n][k] = wo[k][n])
__global__ void cvt_wo_kernel(const float* __restrict__ wo, _Float16* __restrict__ woh) {
  const int n = DD * DD;
  for (int i = blockIdx.x * blockDim.x + threadIdx.x; i < n; i += gridDim.x * blockDim.x) {
    int k = i / DD;
    int c = i % DD;
    woh[(size_t)c * DD + k] = (_Float16)wo[i];
  }
}

// ---------------- stage 1: QKV projection ----------------
// grid (64, H, 3), block 128 (4 waves). Wave computes 16x64 tile of q/k/v for head h.
__global__ void __launch_bounds__(128) qkv_gemm_kernel(
    const _Float16* __restrict__ xh,
    const _Float16* __restrict__ wtq, const _Float16* __restrict__ wtk,
    const _Float16* __restrict__ wtv,
    const float* __restrict__ bq, const float* __restrict__ bk,
    const float* __restrict__ bv,
    _Float16* __restrict__ qh, _Float16* __restrict__ kh,
    _Float16* __restrict__ vth) {
  const int lane = threadIdx.x & 31;
  const int widx = threadIdx.x >> 5;
  const int mt = blockIdx.x * 4 + widx;  // 0..255
  const int h = blockIdx.y;
  const int mat = blockIdx.z;            // 0=Q 1=K 2=V
  const int m0 = mt * 16;
  const int b = m0 >> 10;
  const int s0 = m0 & (SS - 1);

  const _Float16* wt = (mat == 0) ? wtq : (mat == 1) ? wtk : wtv;
  const float* bias = (mat == 0) ? bq : (mat == 1) ? bk : bv;
  const _Float16* wth = wt + (size_t)h * HDm * DD;  // [HD][D]

  v8f acc[4] = {};
  for (int k0 = 0; k0 < DD; k0 += 32) {
    v16h a = load_A16x32(xh + (size_t)m0 * DD + k0, DD, lane);
#pragma unroll
    for (int nt = 0; nt < 4; ++nt) {
      v16h bm = load_BT32x16(wth + (size_t)(nt * 16) * DD + k0, DD, lane);
      acc[nt] = WMMA_F16(a, bm, acc[nt]);
    }
  }

  const int g = lane >> 4, ln = lane & 15;
  const int bh = b * HH + h;
#pragma unroll
  for (int nt = 0; nt < 4; ++nt) {
    const int e = nt * 16 + ln;
    const float bia = bias[h * HDm + e];
    if (mat == 2) {
      v8h pk;  // rows s0+8g .. s0+8g+7 contiguous in Vt[e][s]
#pragma unroll
      for (int r = 0; r < 8; ++r) pk[r] = (_Float16)(acc[nt][r] + bia);
      *(v8h*)(vth + ((size_t)bh * HDm + e) * SS + s0 + 8 * g) = pk;
    } else {
      _Float16* dst = ((mat == 0) ? qh : kh) + (size_t)bh * SS * HDm;
#pragma unroll
      for (int r = 0; r < 8; ++r) {
        const int s = s0 + 8 * g + r;
        dst[(size_t)s * HDm + e] = (_Float16)(acc[nt][r] + bia);
      }
    }
  }
}

// ---------------- stage 2: flash-style sparse attention ----------------
// grid (16, H, B), block 128 (4 waves); wave handles a 16-query tile.
__global__ void __launch_bounds__(128) attn_kernel(
    const _Float16* __restrict__ qh, const _Float16* __restrict__ kh,
    const _Float16* __restrict__ vth, _Float16* __restrict__ ch) {
  const int lane = threadIdx.x & 31;
  const int widx = threadIdx.x >> 5;
  const int h = blockIdx.y;
  const int b = blockIdx.z;
  const int bh = b * HH + h;
  const int i0 = (blockIdx.x * 4 + widx) * 16;

  __shared__ __align__(16) _Float16 Pbuf[4][32 * 16];  // per-wave [key][query] col-major
  _Float16* Pc = Pbuf[widx];

  const _Float16* qb = qh + (size_t)bh * SS * HDm;
  const _Float16* kb = kh + (size_t)bh * SS * HDm;
  const _Float16* vb = vth + (size_t)bh * HDm * SS;

  // Q tile 16x64 resident as two A fragments.
  const v16h qa0 = load_A16x32(qb + (size_t)i0 * HDm + 0, HDm, lane);
  const v16h qa1 = load_A16x32(qb + (size_t)i0 * HDm + 32, HDm, lane);

  const int hp = (h < 4) ? h : 4;
  const int band = 1 << hp;
  const int mm = (2 << hp) - 1;  // diff % 2^(hp+1) == 0  <=>  (diff & mm) == 0

  const int g = lane >> 4, ln = lane & 15;
  const float NEG_INF = -__builtin_inff();

  float mrow[8], lrow[8];
#pragma unroll
  for (int r = 0; r < 8; ++r) { mrow[r] = NEG_INF; lrow[r] = 0.0f; }
  v8f O[4] = {};

  for (int j0 = 0; j0 < SS; j0 += 32) {
    // ---- scores for two 16-key subtiles (K-dim = head dim 64 = 2 wmma steps) ----
    v8f s0 = {}, s1 = {};
    {
      v16h b0 = load_BT32x16(kb + (size_t)j0 * HDm + 0, HDm, lane);
      s0 = WMMA_F16(qa0, b0, s0);
      v16h b1 = load_BT32x16(kb + (size_t)j0 * HDm + 32, HDm, lane);
      s0 = WMMA_F16(qa1, b1, s0);
      v16h b2 = load_BT32x16(kb + (size_t)(j0 + 16) * HDm + 0, HDm, lane);
      s1 = WMMA_F16(qa0, b2, s1);
      v16h b3 = load_BT32x16(kb + (size_t)(j0 + 16) * HDm + 32, HDm, lane);
      s1 = WMMA_F16(qa1, b3, s1);
    }

    const int ja = j0 + ln;        // key col, subtile 0
    const int jb = j0 + 16 + ln;   // key col, subtile 1
    float mnew[8];
#pragma unroll
    for (int r = 0; r < 8; ++r) {
      const int i = i0 + r + 8 * g;  // query row held by this lane for elem r
      float v0 = s0[r] * 0.125f;     // 1/sqrt(64)
      float v1 = s1[r] * 0.125f;
      int d0 = i - ja; d0 = (d0 < 0) ? -d0 : d0;
      int d1 = i - jb; d1 = (d1 < 0) ? -d1 : d1;
      const bool a0 = (d0 <= band) || ((d0 & mm) == 0);
      const bool a1 = (d1 <= band) || ((d1 & mm) == 0);
      v0 = a0 ? v0 : NEG_INF;
      v1 = a1 ? v1 : NEG_INF;
      s0[r] = v0; s1[r] = v1;
      float mx = fmaxf(v0, v1);
      for (int off = 1; off < 16; off <<= 1)  // row max within 16-lane half
        mx = fmaxf(mx, __shfl_xor(mx, off, 32));
      mnew[r] = mx;
    }

    float p0[8], p1[8];
#pragma unroll
    for (int r = 0; r < 8; ++r) {
      const float mp = fmaxf(mrow[r], mnew[r]);
      const float alpha = (mrow[r] == mp) ? 1.0f : __expf(mrow[r] - mp);
      const float q0 = (s0[r] == NEG_INF) ? 0.0f : __expf(s0[r] - mp);
      const float q1 = (s1[r] == NEG_INF) ? 0.0f : __expf(s1[r] - mp);
      float rs = q0 + q1;
      for (int off = 1; off < 16; off <<= 1)  // row sum within half
        rs += __shfl_xor(rs, off, 32);
      lrow[r] = lrow[r] * alpha + rs;
      mrow[r] = mp;
      p0[r] = q0; p1[r] = q1;
#pragma unroll
      for (int nt = 0; nt < 4; ++nt) O[nt][r] *= alpha;
    }

    // ---- P (C-layout f32) -> LDS col-major f16 (one b128 store per subtile) ----
    v8h pk0, pk1;
#pragma unroll
    for (int r = 0; r < 8; ++r) { pk0[r] = (_Float16)p0[r]; pk1[r] = (_Float16)p1[r]; }
    *(v8h*)&Pc[ln * 16 + 8 * g] = pk0;          // Pc[key][query]
    *(v8h*)&Pc[(16 + ln) * 16 + 8 * g] = pk1;

    // ---- read back as A-layout 16x32 fragment ----
    v16h pa;
#pragma unroll
    for (int t = 0; t < 8; ++t) {
      pa[t]     = Pc[(8 * g + t) * 16 + ln];        // K = 8g+t
      pa[8 + t] = Pc[(16 + 8 * g + t) * 16 + ln];   // K = 16+8g+t
    }

    // ---- O += P * V  (B fragments from transposed V: contiguous) ----
#pragma unroll
    for (int nt = 0; nt < 4; ++nt) {
      v16h vm = load_BT32x16(vb + (size_t)(nt * 16) * SS + j0, SS, lane);
      O[nt] = WMMA_F16(pa, vm, O[nt]);
    }
  }

  // ---- epilogue: normalize, store concat buffer [b][s][h*64+e] as f16 ----
  float rinv[8];
#pragma unroll
  for (int r = 0; r < 8; ++r) rinv[r] = (lrow[r] > 0.0f) ? 1.0f / lrow[r] : 0.0f;
  _Float16* cb = ch + (size_t)b * SS * DD;
#pragma unroll
  for (int nt = 0; nt < 4; ++nt) {
    const int e = nt * 16 + ln;
#pragma unroll
    for (int r = 0; r < 8; ++r) {
      const int s = i0 + 8 * g + r;
      cb[(size_t)s * DD + h * HDm + e] = (_Float16)(O[nt][r] * rinv[r]);
    }
  }
}

// ---------------- stage 3: output projection ----------------
// grid (64, 8), block 128 (4 waves). Wave computes 16x64 tile of [4096,512] output.
__global__ void __launch_bounds__(128) out_gemm_kernel(
    const _Float16* __restrict__ ch, const _Float16* __restrict__ woh,
    const float* __restrict__ bo, float* __restrict__ out) {
  const int lane = threadIdx.x & 31;
  const int widx = threadIdx.x >> 5;
  const int m0 = (blockIdx.x * 4 + widx) * 16;
  const int n0 = blockIdx.y * 64;

  v8f acc[4] = {};
  for (int k0 = 0; k0 < DD; k0 += 32) {
    v16h a = load_A16x32(ch + (size_t)m0 * DD + k0, DD, lane);
#pragma unroll
    for (int nt = 0; nt < 4; ++nt) {
      v16h bm = load_BT32x16(woh + (size_t)(n0 + nt * 16) * DD + k0, DD, lane);
      acc[nt] = WMMA_F16(a, bm, acc[nt]);
    }
  }

  const int g = lane >> 4, ln = lane & 15;
#pragma unroll
  for (int nt = 0; nt < 4; ++nt) {
    const int col = n0 + nt * 16 + ln;
    const float bb = bo[col];
#pragma unroll
    for (int r = 0; r < 8; ++r) {
      out[(size_t)(m0 + 8 * g + r) * DD + col] = acc[nt][r] + bb;
    }
  }
}

// ---------------- launch ----------------
extern "C" void kernel_launch(void* const* d_in, const int* in_sizes, int n_in,
                              void* d_out, int out_size, void* d_ws, size_t ws_size,
                              hipStream_t stream) {
  (void)in_sizes; (void)n_in; (void)out_size; (void)ws_size;
  const float* x  = (const float*)d_in[0];
  const float* Wq = (const float*)d_in[1];
  const float* bq = (const float*)d_in[2];
  const float* Wk = (const float*)d_in[3];
  const float* bk = (const float*)d_in[4];
  const float* Wv = (const float*)d_in[5];
  const float* bv = (const float*)d_in[6];
  const float* Wo = (const float*)d_in[7];
  const float* bo = (const float*)d_in[8];
  float* out = (float*)d_out;

  const size_t NX  = (size_t)BB * SS * DD;       // 2,097,152
  const size_t NW  = (size_t)HH * HDm * DD;      //   262,144
  const size_t NWO = (size_t)DD * DD;            //   262,144
  const size_t NQ  = (size_t)BB * HH * SS * HDm; // 2,097,152

  _Float16* xh  = (_Float16*)d_ws;
  _Float16* wtq = xh + NX;
  _Float16* wtk = wtq + NW;
  _Float16* wtv = wtk + NW;
  _Float16* woh = wtv + NW;
  _Float16* qh  = woh + NWO;
  _Float16* kh  = qh + NQ;
  _Float16* vth = kh + NQ;
  _Float16* chb = vth + NQ;  // concat attention output, f16 [B*S][D]

  cvt_x_kernel<<<1024, 256, 0, stream>>>(x, xh, (int)NX);
  cvt_w_kernel<<<256, 256, 0, stream>>>(Wq, wtq);
  cvt_w_kernel<<<256, 256, 0, stream>>>(Wk, wtk);
  cvt_w_kernel<<<256, 256, 0, stream>>>(Wv, wtv);
  cvt_wo_kernel<<<256, 256, 0, stream>>>(Wo, woh);

  qkv_gemm_kernel<<<dim3(64, HH, 3), 128, 0, stream>>>(
      xh, wtq, wtk, wtv, bq, bk, bv, qh, kh, vth);

  attn_kernel<<<dim3(16, HH, BB), 128, 0, stream>>>(qh, kh, vth, chb);

  out_gemm_kernel<<<dim3(64, 8), 128, 0, stream>>>(chb, woh, bo, out);
}
